// SSSTProSurrogate_89799176225412
// MI455X (gfx1250) — compile-verified, matrix operand order checked by reading, not run on hardware
//
#include <hip/hip_runtime.h>
#include <hip/hip_bf16.h>
#include <math.h>

// ---------------------------------------------------------------------------
// Types for CDNA5 WMMA (wave32): A/B operands are v16h (8 VGPRs of packed f16),
// C/D accumulators are v8f (8 VGPRs of f32).
// ---------------------------------------------------------------------------
typedef _Float16 v16h __attribute__((ext_vector_type(16)));
typedef _Float16 v8h  __attribute__((ext_vector_type(8)));
typedef float    v8f  __attribute__((ext_vector_type(8)));

union V16 { v16h v; v8h h[2]; };

static constexpr int BB = 8, TT = 1024, DD = 256, HH = 8, DH = 32;
static constexpr int NBLK = 4, NE = 8;

__device__ __forceinline__ float gelu_f(float x) {
  // exact erf GELU (torch default)
  return 0.5f * x * (1.0f + erff(x * 0.70710678118654752f));
}

__device__ __forceinline__ v8f wmma16(const V16 &a, const V16 &b, v8f c) {
  return __builtin_amdgcn_wmma_f32_16x16x32_f16(false, a.v, false, b.v,
                                                (short)0, c, false, false);
}

__device__ __forceinline__ float wave_sum32(float v) {
#pragma unroll
  for (int off = 1; off < 32; off <<= 1) v += __shfl_xor(v, off, 32);
  return v;
}
__device__ __forceinline__ float half_sum16(float v) {
#pragma unroll
  for (int off = 1; off < 16; off <<= 1) v += __shfl_xor(v, off, 32);
  return v;
}
__device__ __forceinline__ float half_max16(float v) {
#pragma unroll
  for (int off = 1; off < 16; off <<= 1) v = fmaxf(v, __shfl_xor(v, off, 32));
  return v;
}

// ---------------------------------------------------------------------------
// WMMA GEMM:  C[M,N] = act( A[M,K] * W[N,K]^T + bias[N] )
// 128 threads (4 waves), 128x64 block tile; each wave owns a 32x64 strip
// (2 A operands x 4 B tiles = 8 WMMAs per K-step). K stepped by 32 through
// LDS (f32 -> f16 at staging). M multiple of 128, N of 64, K of 32.
// Prefetches are unconditional: default-TH prefetch is speculative, so
// past-the-end addresses are silently dropped (ISA 10.5) -- no branches.
// ---------------------------------------------------------------------------
__global__ __launch_bounds__(128) void k_gemm(const float *__restrict__ A,
                                              const float *__restrict__ W,
                                              const float *__restrict__ bias,
                                              float *__restrict__ C,
                                              int M, int N, int K, int act) {
  __shared__ __align__(16) _Float16 As[128][40]; // 80B row stride (16B aligned)
  __shared__ __align__(16) _Float16 Ws[64][40];
  const int tiles_n = N >> 6;
  const int m0 = (blockIdx.x / tiles_n) << 7;
  const int n0 = (blockIdx.x % tiles_n) << 6;
  const int tid = threadIdx.x;
  const int wv = tid >> 5, lane = tid & 31;
  const int ml = lane & 15;
  const int hi = (lane >> 4) & 1;
  const int kb0 = hi ? 8 : 0;

  v8f acc[2][4];
#pragma unroll
  for (int s = 0; s < 2; ++s)
#pragma unroll
    for (int j = 0; j < 4; ++j)
#pragma unroll
      for (int r = 0; r < 8; ++r) acc[s][j][r] = 0.0f;

  for (int k0 = 0; k0 < K; k0 += 32) {
    // stage A tile (128x32) as f16 -- 1024 float4 loads
#pragma unroll
    for (int i = 0; i < 8; ++i) {
      int idx = tid + i * 128;
      int r = idx >> 3, c = (idx & 7) << 2;
      const float *ap = A + (size_t)(m0 + r) * K + k0 + c;
      float4 fa = *(const float4 *)ap;
      __builtin_prefetch((const void *)(ap + 32), 0, 1); // speculative, branch-free
      As[r][c + 0] = (_Float16)fa.x; As[r][c + 1] = (_Float16)fa.y;
      As[r][c + 2] = (_Float16)fa.z; As[r][c + 3] = (_Float16)fa.w;
    }
    // stage W tile (64x32) as f16 -- 512 float4 loads
#pragma unroll
    for (int i = 0; i < 4; ++i) {
      int idx = tid + i * 128;
      int r = idx >> 3, c = (idx & 7) << 2;
      const float *wp = W + (size_t)(n0 + r) * K + k0 + c;
      float4 fw = *(const float4 *)wp;
      __builtin_prefetch((const void *)(wp + 32), 0, 1);
      Ws[r][c + 0] = (_Float16)fw.x; Ws[r][c + 1] = (_Float16)fw.y;
      Ws[r][c + 2] = (_Float16)fw.z; Ws[r][c + 3] = (_Float16)fw.w;
    }
    __syncthreads();
    V16 a0, a1;
    a0.h[0] = *(const v8h *)(&As[wv * 32 + ml][kb0]);
    a0.h[1] = *(const v8h *)(&As[wv * 32 + ml][kb0 + 16]);
    a1.h[0] = *(const v8h *)(&As[wv * 32 + 16 + ml][kb0]);
    a1.h[1] = *(const v8h *)(&As[wv * 32 + 16 + ml][kb0 + 16]);
#pragma unroll
    for (int j = 0; j < 4; ++j) {
      V16 b;
      b.h[0] = *(const v8h *)(&Ws[j * 16 + ml][kb0]);
      b.h[1] = *(const v8h *)(&Ws[j * 16 + ml][kb0 + 16]);
      acc[0][j] = wmma16(a0, b, acc[0][j]);
      acc[1][j] = wmma16(a1, b, acc[1][j]);
    }
    __syncthreads();
  }
#pragma unroll
  for (int s = 0; s < 2; ++s)
#pragma unroll
    for (int j = 0; j < 4; ++j) {
      int n = n0 + j * 16 + ml;
      float bv = bias[n];
#pragma unroll
      for (int r = 0; r < 8; ++r) {
        int m = m0 + wv * 32 + s * 16 + r + hi * 8;
        float v = acc[s][j][r] + bv;
        if (act) v = gelu_f(v);
        C[(size_t)m * N + n] = v;
      }
    }
}

// ---------------------------------------------------------------------------
// Flash attention, dh = 32. qkv layout (B,T,3D); out (B,T,D).
// 4 waves / block, each wave owns a 16-row query strip; online softmax.
// K tile (key-major) and V tile (dh-major / transposed) are cooperatively
// staged into LDS as f16 so every WMMA operand is two ds_load_b128s.
// grid = B*H*(T/64)
// ---------------------------------------------------------------------------
__global__ __launch_bounds__(128) void k_attn(const float *__restrict__ qkv,
                                              float *__restrict__ out) {
  const int blk = blockIdx.x;
  const int tb = blk & 15;        // T/64 = 16 row-blocks
  const int h = (blk >> 4) & 7;
  const int b = blk >> 7;
  const int tid = threadIdx.x;
  const int wv = tid >> 5, lane = tid & 31;
  const int ml = lane & 15;
  const int hi = (lane >> 4) & 1;
  const int kb0 = hi ? 8 : 0;
  const int t0 = tb * 64 + wv * 16;

  __shared__ __align__(16) _Float16 Kt[32][40];      // [key][dh]
  __shared__ __align__(16) _Float16 Vt[32][40];      // [dh][key] (transposed)
  __shared__ __align__(16) _Float16 P[4][16][32];    // per-wave P transpose pad

  const float scale = 0.17677669529663687f; // 1/sqrt(32)
  V16 qa;
  {
    const float *qp = qkv + ((size_t)(b * TT + t0 + ml) * 768) + h * DH;
#pragma unroll
    for (int e = 0; e < 16; ++e) {
      int kk = (e < 8 ? e : e + 8) + kb0;
      qa.v[e] = (_Float16)(qp[kk] * scale);
    }
  }
  v8f o0, o1;
  float rmax[8], rsum[8];
#pragma unroll
  for (int r = 0; r < 8; ++r) { o0[r] = 0.f; o1[r] = 0.f; rmax[r] = -3.0e38f; rsum[r] = 0.f; }

  const int kr = tid >> 2;        // staging: key row 0..31
  const int c8 = (tid & 3) * 8;   // staging: dh column base 0,8,16,24

  for (int kb = 0; kb < TT; kb += 32) {
    // ---- cooperative staging of K (key-major) and V (dh-major) as f16 ----
    {
      const float *kg = qkv + (size_t)(b * TT + kb + kr) * 768 + DD + h * DH + c8;
      float4 f0 = *(const float4 *)kg;
      float4 f1 = *(const float4 *)(kg + 4);
      __builtin_prefetch((const void *)(kg + 32 * 768), 0, 1); // speculative
      Kt[kr][c8 + 0] = (_Float16)f0.x; Kt[kr][c8 + 1] = (_Float16)f0.y;
      Kt[kr][c8 + 2] = (_Float16)f0.z; Kt[kr][c8 + 3] = (_Float16)f0.w;
      Kt[kr][c8 + 4] = (_Float16)f1.x; Kt[kr][c8 + 5] = (_Float16)f1.y;
      Kt[kr][c8 + 6] = (_Float16)f1.z; Kt[kr][c8 + 7] = (_Float16)f1.w;
      const float *vg = qkv + (size_t)(b * TT + kb + kr) * 768 + 2 * DD + h * DH + c8;
      float4 g0 = *(const float4 *)vg;
      float4 g1 = *(const float4 *)(vg + 4);
      Vt[c8 + 0][kr] = (_Float16)g0.x; Vt[c8 + 1][kr] = (_Float16)g0.y;
      Vt[c8 + 2][kr] = (_Float16)g0.z; Vt[c8 + 3][kr] = (_Float16)g0.w;
      Vt[c8 + 4][kr] = (_Float16)g1.x; Vt[c8 + 5][kr] = (_Float16)g1.y;
      Vt[c8 + 6][kr] = (_Float16)g1.z; Vt[c8 + 7][kr] = (_Float16)g1.w;
    }
    __syncthreads();

    // ---- S = Q * K^T for 32 keys (two WMMAs) ----
    V16 kop0, kop1;
    kop0.h[0] = *(const v8h *)(&Kt[ml][kb0]);
    kop0.h[1] = *(const v8h *)(&Kt[ml][kb0 + 16]);
    kop1.h[0] = *(const v8h *)(&Kt[16 + ml][kb0]);
    kop1.h[1] = *(const v8h *)(&Kt[16 + ml][kb0 + 16]);
    v8f z;
#pragma unroll
    for (int r = 0; r < 8; ++r) z[r] = 0.f;
    v8f s0 = wmma16(qa, kop0, z);
    v8f s1 = wmma16(qa, kop1, z);

    // ---- online softmax per row (rows live per-VGPR within a 16-lane half)
#pragma unroll
    for (int r = 0; r < 8; ++r) {
      float mx = half_max16(fmaxf(s0[r], s1[r]));
      float nm = fmaxf(rmax[r], mx);
      float alpha = expf(rmax[r] - nm);
      rmax[r] = nm;
      float p0 = expf(s0[r] - nm);
      float p1 = expf(s1[r] - nm);
      s0[r] = p0; s1[r] = p1;
      rsum[r] = rsum[r] * alpha + half_sum16(p0 + p1);
      o0[r] *= alpha; o1[r] *= alpha;
    }
    // ---- transpose P (C layout) -> A-operand layout via per-wave LDS ----
#pragma unroll
    for (int r = 0; r < 8; ++r) {
      P[wv][r + hi * 8][ml] = (_Float16)s0[r];
      P[wv][r + hi * 8][16 + ml] = (_Float16)s1[r];
    }
    V16 pa;
    pa.h[0] = *(const v8h *)(&P[wv][ml][kb0]);
    pa.h[1] = *(const v8h *)(&P[wv][ml][kb0 + 16]);
    // ---- O += P * V (two WMMAs, dh split 16+16) ----
    V16 vb0, vb1;
    vb0.h[0] = *(const v8h *)(&Vt[ml][kb0]);
    vb0.h[1] = *(const v8h *)(&Vt[ml][kb0 + 16]);
    vb1.h[0] = *(const v8h *)(&Vt[16 + ml][kb0]);
    vb1.h[1] = *(const v8h *)(&Vt[16 + ml][kb0 + 16]);
    o0 = wmma16(pa, vb0, o0);
    o1 = wmma16(pa, vb1, o1);
    __syncthreads(); // before next staging overwrites Kt/Vt
  }
#pragma unroll
  for (int r = 0; r < 8; ++r) {
    int t = t0 + r + hi * 8;
    float inv = 1.0f / rsum[r];
    float *op = out + (size_t)(b * TT + t) * DD + h * DH;
    op[ml] = o0[r] * inv;
    op[16 + ml] = o1[r] * inv;
  }
}

// ---------------------------------------------------------------------------
// LayerNorm over D=256 with optional residual: out = LN(x [+ res]) * g + b
// one wave per row; grid = rows/8, block = 256.
// ---------------------------------------------------------------------------
__global__ __launch_bounds__(256) void k_ln(const float *__restrict__ x,
                                            const float *__restrict__ res,
                                            const float *__restrict__ g,
                                            const float *__restrict__ bvec,
                                            float *__restrict__ out) {
  const int row = blockIdx.x * 8 + (threadIdx.x >> 5);
  const int lane = threadIdx.x & 31;
  const float *xp = x + (size_t)row * DD;
  float v[8];
#pragma unroll
  for (int i = 0; i < 2; ++i) {
    int d = lane * 4 + i * 128;
    float4 a = *(const float4 *)(xp + d);
    if (res) {
      float4 r = *(const float4 *)(res + (size_t)row * DD + d);
      a.x += r.x; a.y += r.y; a.z += r.z; a.w += r.w;
    }
    v[i * 4 + 0] = a.x; v[i * 4 + 1] = a.y; v[i * 4 + 2] = a.z; v[i * 4 + 3] = a.w;
  }
  float s = 0.f, s2 = 0.f;
#pragma unroll
  for (int k = 0; k < 8; ++k) { s += v[k]; s2 += v[k] * v[k]; }
  s = wave_sum32(s); s2 = wave_sum32(s2);
  float m = s * (1.0f / 256.0f);
  float var = s2 * (1.0f / 256.0f) - m * m;
  float rinv = rsqrtf(var + 1e-5f);
#pragma unroll
  for (int i = 0; i < 2; ++i)
#pragma unroll
    for (int c = 0; c < 4; ++c) {
      int d = lane * 4 + i * 128 + c;
      out[(size_t)row * DD + d] = (v[i * 4 + c] - m) * rinv * g[d] + bvec[d];
    }
}

// ctx projection: ctxp[b,d] = b[d] + sum_k W[d,k]*ctx[b,k]   (K=24)
__global__ void k_ctxproj(const float *__restrict__ ctx, const float *__restrict__ w,
                          const float *__restrict__ bias, float *__restrict__ ctxp) {
  int b = blockIdx.x, d = threadIdx.x;
  float a = bias[d];
  for (int k = 0; k < 24; ++k) a += w[d * 24 + k] * ctx[b * 24 + k];
  ctxp[b * DD + d] = a;
}

// s embedding: per (b,t): 1 -> 128 (gelu) -> 256, + ctx projection
__global__ __launch_bounds__(128) void k_sembed(const float *__restrict__ s,
                                                const float *__restrict__ w1,
                                                const float *__restrict__ b1,
                                                const float *__restrict__ w2,
                                                const float *__restrict__ b2,
                                                const float *__restrict__ ctxp,
                                                float *__restrict__ h) {
  int row = blockIdx.x;       // b*T + t
  int b = row >> 10;
  int j = threadIdx.x;
  __shared__ float hid[128];
  float sv = s[row];
  hid[j] = gelu_f(w1[j] * sv + b1[j]);
  __syncthreads();
#pragma unroll
  for (int p = 0; p < 2; ++p) {
    int d = j + p * 128;
    float a = b2[d] + ctxp[b * DD + d];
    for (int k = 0; k < 128; ++k) a += w2[d * 128 + k] * hid[k];
    h[(size_t)row * DD + d] = a;
  }
}

// grouped trunk conv: K=5 pad 2, groups=32 (8 ch/group); h,out in (B,T,D)
__global__ void k_groupconv(const float *__restrict__ h, const float *__restrict__ w,
                            const float *__restrict__ bias, float *__restrict__ out) {
  int idx = blockIdx.x * blockDim.x + threadIdx.x; // B*T*D
  int d = idx & 255, t = (idx >> 8) & 1023, b = idx >> 18;
  int g = d >> 3;
  float a = bias[d];
#pragma unroll
  for (int tap = 0; tap < 5; ++tap) {
    int tt = t + tap - 2;
    if (tt < 0 || tt >= TT) continue;
    const float *hp = h + (size_t)(b * TT + tt) * DD + g * 8;
#pragma unroll
    for (int ic = 0; ic < 8; ++ic) a += hp[ic] * w[(d * 8 + ic) * 5 + tap];
  }
  out[idx] = a;
}

// expert depthwise dilated conv + GELU
__global__ void k_dw(const float *__restrict__ h, const float *__restrict__ w,
                     const float *__restrict__ bias, float *__restrict__ out,
                     int ks, int dil) {
  int idx = blockIdx.x * blockDim.x + threadIdx.x;
  int d = idx & 255, t = (idx >> 8) & 1023, b = idx >> 18;
  int pad = (ks - 1) * dil / 2;
  float a = bias[d];
  for (int tap = 0; tap < ks; ++tap) {
    int tt = t + tap * dil - pad;
    if (tt >= 0 && tt < TT) a += h[(size_t)(b * TT + tt) * DD + d] * w[d * ks + tap];
  }
  out[idx] = gelu_f(a);
}

// GroupNorm(8) over (C/G=32, T) + weighted accumulate into moe buffer
__global__ __launch_bounds__(256) void k_gn_accum(const float *__restrict__ x,
                                                  const float *__restrict__ gg,
                                                  const float *__restrict__ gb,
                                                  const float *__restrict__ we,
                                                  int e, float *__restrict__ moe) {
  int b = blockIdx.x >> 3, grp = blockIdx.x & 7;
  int c0 = grp * 32;
  int tid = threadIdx.x, wv = tid >> 5, lane = tid & 31;
  float s = 0.f, s2 = 0.f;
  for (int i = tid; i < 32768; i += 256) {
    int t = i >> 5, c = i & 31;
    float v = x[(size_t)(b * TT + t) * DD + c0 + c];
    s += v; s2 += v * v;
  }
  __shared__ float ps[8], ps2[8], sm[2];
  s = wave_sum32(s); s2 = wave_sum32(s2);
  if (lane == 0) { ps[wv] = s; ps2[wv] = s2; }
  __syncthreads();
  if (tid == 0) {
    float a = 0.f, a2 = 0.f;
    for (int w2i = 0; w2i < 8; ++w2i) { a += ps[w2i]; a2 += ps2[w2i]; }
    float m = a * (1.0f / 32768.0f);
    float var = a2 * (1.0f / 32768.0f) - m * m;
    sm[0] = m; sm[1] = rsqrtf(var + 1e-5f);
  }
  __syncthreads();
  float m = sm[0], rinv = sm[1];
  float wgt = we[b * 8 + e];
  for (int i = tid; i < 32768; i += 256) {
    int t = i >> 5, c = i & 31;
    size_t idx = (size_t)(b * TT + t) * DD + c0 + c;
    moe[idx] += wgt * ((x[idx] - m) * rinv * gg[c0 + c] + gb[c0 + c]);
  }
}

__global__ void k_zero(float *__restrict__ p, int n) {
  int i = blockIdx.x * blockDim.x + threadIdx.x;
  if (i < n) p[i] = 0.f;
}
__global__ void k_addmoe(float *__restrict__ h, const float *__restrict__ moe) {
  int i = blockIdx.x * blockDim.x + threadIdx.x;
  h[i] += moe[i];
}

// mean over T: pooled[b,d]
__global__ void k_pool(const float *__restrict__ h, float *__restrict__ pooled) {
  int idx = blockIdx.x * blockDim.x + threadIdx.x; // B*D
  int d = idx & 255, b = idx >> 8;
  float s = 0.f;
  for (int t = 0; t < TT; ++t) s += h[(size_t)(b * TT + t) * DD + d];
  pooled[idx] = s * (1.0f / (float)TT);
}

// small 2-layer MLP per batch row: K -> Hh (gelu) -> O
__global__ void k_smlp(const float *__restrict__ x, const float *__restrict__ w1,
                       const float *__restrict__ b1, const float *__restrict__ w2,
                       const float *__restrict__ b2, float *__restrict__ out,
                       int K, int Hh, int O) {
  int b = blockIdx.x, j = threadIdx.x;
  __shared__ float hid[256];
  if (j < Hh) {
    float a = b1[j];
    for (int k = 0; k < K; ++k) a += x[b * K + k] * w1[j * K + k];
    hid[j] = gelu_f(a);
  }
  __syncthreads();
  if (j < O) {
    float a = b2[j];
    for (int k = 0; k < Hh; ++k) a += hid[k] * w2[j * Hh + k];
    out[b * O + j] = a;
  }
}

// top-2 of 8 logits + softmax; writes weights (B,2) and dense w_e (B,8)
__global__ void k_top2(const float *__restrict__ logits, float *__restrict__ wout,
                       float *__restrict__ we) {
  int b = threadIdx.x;
  if (b >= BB) return;
  float v[8];
  for (int e = 0; e < 8; ++e) v[e] = logits[b * 8 + e];
  int i1 = 0;
  for (int e = 1; e < 8; ++e) if (v[e] > v[i1]) i1 = e;
  float v1 = v[i1];
  v[i1] = -3.0e38f;
  int i2 = 0;
  for (int e = 1; e < 8; ++e) if (v[e] > v[i2]) i2 = e;
  float v2 = v[i2];
  float e2 = expf(v2 - v1);
  float inv = 1.0f / (1.0f + e2);
  float w1 = inv, w2 = e2 * inv;
  wout[b * 2 + 0] = w1;
  wout[b * 2 + 1] = w2;
  for (int e = 0; e < 8; ++e) we[b * 8 + e] = 0.f;
  we[b * 8 + i1] += w1;
  we[b * 8 + i2] += w2;
}

// 128-wide dot per row: out[r] = bias + sum_k hid[r,k]*w[k]
__global__ void k_dot128(const float *__restrict__ hid, const float *__restrict__ w,
                         const float *__restrict__ bias, float *__restrict__ out, int n) {
  int r = blockIdx.x * blockDim.x + threadIdx.x;
  if (r >= n) return;
  float a = bias[0];
  for (int k = 0; k < 128; ++k) a += hid[(size_t)r * 128 + k] * w[k];
  out[r] = a;
}

// sw head conv1: (128,256,3) pad 1 on (B,T,D) layout, fused GELU
__global__ void k_swconv(const float *__restrict__ h, const float *__restrict__ w,
                         const float *__restrict__ bias, float *__restrict__ out) {
  int idx = blockIdx.x * blockDim.x + threadIdx.x; // B*T*128
  int o = idx & 127, t = (idx >> 7) & 1023, b = idx >> 17;
  float a = bias[o];
#pragma unroll
  for (int tap = 0; tap < 3; ++tap) {
    int tt = t + tap - 1;
    if (tt < 0 || tt >= TT) continue;
    const float *hp = h + (size_t)(b * TT + tt) * DD;
    const float *wp = w + (size_t)o * 256 * 3 + tap; // stride 3 over ic
    for (int ic = 0; ic < 256; ++ic) a += hp[ic] * wp[ic * 3];
  }
  out[idx] = gelu_f(a);
}

// scatter sc (B,2) into d_out[16384..16392) and [16392..16400)
__global__ void k_scfin(const float *__restrict__ sc, float *__restrict__ out) {
  int b = threadIdx.x;
  if (b >= BB) return;
  out[16384 + b] = sc[b * 2 + 0];
  out[16392 + b] = sc[b * 2 + 1];
}

// ---------------------------------------------------------------------------
extern "C" void kernel_launch(void* const* d_in, const int* in_sizes, int n_in,
                              void* d_out, int out_size, void* d_ws, size_t ws_size,
                              hipStream_t stream) {
  (void)in_sizes; (void)n_in; (void)out_size; (void)ws_size;
  auto F = [&](int i) { return (const float *)d_in[i]; };
  float *ws = (float *)d_ws;
  float *out = (float *)d_out;

  // workspace layout (floats)
  float *h      = ws + 0;          // B*T*D = 2,097,152
  float *U      = ws + 2097152;    // shared region: qkv (6.29M) / ff hidden (8.39M) / head hiddens
  float *t1     = ws + 10485760;   // B*T*D
  float *t2     = ws + 12582912;   // B*T*D
  float *moe    = ws + 14680064;   // B*T*D
  float *pooled = ws + 16777216;   // B*D
  float *ctxp   = ws + 16779264;   // B*D
  float *glog   = ws + 16781312;   // B*8
  float *we     = ws + 16781376;   // B*8
  float *scbuf  = ws + 16781440;   // B*2

  const int KS[8]  = {3, 5, 7, 9, 3, 5, 7, 9};
  const int DIL[8] = {1, 1, 1, 1, 2, 2, 2, 2};
  const int ROWS = BB * TT; // 8192

  // input pytree order (jax sorted-key flatten): 0 ctx, 1 s, then params
  k_ctxproj<<<BB, 256, 0, stream>>>(F(0), F(3), F(2), ctxp);
  k_sembed<<<ROWS, 128, 0, stream>>>(F(1), F(71), F(70), F(73), F(72), ctxp, h);

  for (int i = 0; i < NBLK; ++i) {
    int p = 74 + i * 16; // b_in,b_out,w_in,w_out, conv.b, conv.w, ff1.b,ff1.w, ff2.b,ff2.w, ln1.b,g, ln2.b,g, ln3.b,g
    k_groupconv<<<(ROWS * DD) / 256, 256, 0, stream>>>(h, F(p + 5), F(p + 4), t1);
    k_ln<<<ROWS / 8, 256, 0, stream>>>(h, t1, F(p + 11), F(p + 10), h);
    k_gemm<<<(ROWS / 128) * (768 / 64), 128, 0, stream>>>(h, F(p + 2), F(p + 0), U, ROWS, 768, 256, 0);
    k_attn<<<BB * HH * (TT / 64), 128, 0, stream>>>(U, t1);
    k_gemm<<<(ROWS / 128) * (256 / 64), 128, 0, stream>>>(t1, F(p + 3), F(p + 1), t2, ROWS, 256, 256, 0);
    k_ln<<<ROWS / 8, 256, 0, stream>>>(h, t2, F(p + 13), F(p + 12), h);
    k_gemm<<<(ROWS / 128) * (1024 / 64), 128, 0, stream>>>(h, F(p + 7), F(p + 6), U, ROWS, 1024, 256, 1);
    k_gemm<<<(ROWS / 128) * (256 / 64), 128, 0, stream>>>(U, F(p + 9), F(p + 8), t1, ROWS, 256, 1024, 0);
    k_ln<<<ROWS / 8, 256, 0, stream>>>(h, t1, F(p + 15), F(p + 14), h);
  }

  // gating on pre-MoE pooled features
  k_pool<<<(BB * DD) / 256, 256, 0, stream>>>(h, pooled);
  k_smlp<<<BB, 256, 0, stream>>>(pooled, F(53), F(52), F(55), F(54), glog, 256, 256, 8);
  k_top2<<<1, 32, 0, stream>>>(glog, out + 16400, we);

  // MoE: dense loop over all 8 experts with per-(b,e) scalar weights
  k_zero<<<(ROWS * DD) / 256, 256, 0, stream>>>(moe, ROWS * DD);
  for (int e = 0; e < NE; ++e) {
    int q = 4 + e * 6; // dw.b, dw.w, gn.b, gn.g, pw.b, pw.w
    k_dw<<<(ROWS * DD) / 256, 256, 0, stream>>>(h, F(q + 1), F(q + 0), t1, KS[e], DIL[e]);
    k_gemm<<<(ROWS / 128) * (256 / 64), 128, 0, stream>>>(t1, F(q + 5), F(q + 4), t2, ROWS, 256, 256, 0);
    k_gn_accum<<<BB * 8, 256, 0, stream>>>(t2, F(q + 3), F(q + 2), we, e, moe);
  }
  k_addmoe<<<(ROWS * DD) / 256, 256, 0, stream>>>(h, moe);

  // head_main: LN -> 256->128 gelu -> 128->1
  k_ln<<<ROWS / 8, 256, 0, stream>>>(h, nullptr, F(61), F(60), t2);
  k_gemm<<<(ROWS / 128) * (128 / 64), 128, 0, stream>>>(t2, F(57), F(56), U, ROWS, 128, 256, 1);
  k_dot128<<<ROWS / 256, 256, 0, stream>>>(U, F(59), F(58), out + 0, ROWS);

  // head_sw_pt: conv(256->128,k3) gelu -> conv(128->1,k1)
  k_swconv<<<(ROWS * 128) / 256, 256, 0, stream>>>(h, F(63), F(62), U);
  k_dot128<<<ROWS / 256, 256, 0, stream>>>(U, F(65), F(64), out + ROWS, ROWS);

  // head_sw_sc: pooled -> 128 gelu -> 2
  k_smlp<<<BB, 128, 0, stream>>>(pooled, F(67), F(66), F(69), F(68), scbuf, 256, 128, 2);
  k_scfin<<<1, 32, 0, stream>>>(scbuf, out);
}
